// Generator_2_52621939310880
// MI455X (gfx1250) — matrix operand order repaired and checked by reading, not compiled, not run on hardware
//
#include <hip/hip_runtime.h>
#include <stdint.h>

// ---------------------------------------------------------------------------
// CDNA5 (gfx1250) wave32 WMMA implementation of the generator forward pass.
// Convs = implicit GEMM with v_wmma_f32_16x16x32_f16 (f16 operands, f32 acc).
// v5: weight tile staged via GLOBAL_LOAD_ASYNC_TO_LDS (ASYNCcnt path) with
//     the toolchain's AS(1)/AS(3) int* signature; sync fallback otherwise.
// ---------------------------------------------------------------------------

typedef __attribute__((ext_vector_type(16))) _Float16 v16h;
typedef __attribute__((ext_vector_type(8)))  float    v8f;
typedef __attribute__((ext_vector_type(4)))  float    v4f;

#define BATCH 64
#define BN_EPS 1e-5f

#ifndef __has_builtin
#define __has_builtin(x) 0
#endif
#if __has_builtin(__builtin_amdgcn_global_load_async_to_lds_b32) && \
    __has_builtin(__builtin_amdgcn_s_wait_asynccnt)
#define USE_ASYNC_LDS 1
typedef __attribute__((address_space(1))) int as1_int;
typedef __attribute__((address_space(3))) int as3_int;
#else
#define USE_ASYNC_LDS 0
#endif

__device__ __forceinline__ unsigned hash_u32(unsigned x) {
    x ^= x >> 16; x *= 0x7feb352dU;
    x ^= x >> 15; x *= 0x846ca68bU;
    x ^= x >> 16;
    return x;
}

// ---------------------------------------------------------------------------
// Conv2d (NCHW, OIHW) as implicit GEMM:
//   D[Cout x P] = W[Cout x Cin*K*K] * Patch[Cin*K*K x P],   P = B*OH*OW
// Block = 128 threads = 4 waves; each wave owns a 16x16 tile (same M tile,
// four adjacent N tiles). Weight tile staged once per block in LDS (f32,
// via async-to-LDS when available; converted to f16 at fragment build).
// A-fragment layout (16-bit): lanes 0-15 hold K {0..7,16..23}, lanes 16-31
// hold K {8..15,24..31}.
// ---------------------------------------------------------------------------
template<int KSZ>
__global__ void conv2d_wmma_kernel(const float* __restrict__ x, const float* __restrict__ w,
                                   const float* __restrict__ bias, float* __restrict__ y,
                                   int Cin, int H, int W, int Cout, int OH, int OW,
                                   int pad) {
    __shared__ __align__(16) float Ashf[16][32];   // [m][k] tile, 2 KB (f32)

    const int tid   = threadIdx.x;
    const int lane  = tid & 31;
    const int wave  = tid >> 5;
    const int hi    = lane >> 4;
    const int lo    = lane & 15;
    const int mBase = blockIdx.y * 16;
    const int nTile = blockIdx.x * 4 + wave;
    const int P     = BATCH * OH * OW;
    const int KK    = Cin * KSZ * KSZ;
    const int OHW   = OH * OW;

    // decode the output pixel this lane's B-column corresponds to (clamped)
    const int  col  = nTile * 16 + lo;
    const bool colv = (col < P);
    const int  cc   = colv ? col : 0;
    const int  bN   = cc / OHW;
    const int  rP   = cc - bN * OHW;
    const int  oh   = rP / OW;
    const int  ow   = rP - oh * OW;
    const int  xbase = bN * Cin;                // channel-plane base for this pixel

    // A-staging: thread (tid) covers 4 consecutive k for row am of the tile
    const int am = tid >> 3;                    // 0..15
    const int ak = (tid & 7) * 4;               // 0,4,...,28
    const int grow   = mBase + am;
    const bool rowv  = (grow < Cout);
    const unsigned wrow = (unsigned)(rowv ? grow : 0) * (unsigned)KK;

    v8f acc = {};
    for (int kk = 0; kk < KK; kk += 32) {
        const bool chunkFull = rowv && (kk + ak + 4 <= KK);   // all 4 slots valid
#if USE_ASYNC_LDS
        // ---- async global->LDS weight staging (no VGPR round trip) ----
#pragma unroll
        for (int t = 0; t < 4; ++t) {
            int k = kk + ak + t;
            if (rowv && k < KK)
                __builtin_amdgcn_global_load_async_to_lds_b32(
                    (as1_int*)(w + wrow + (unsigned)k),
                    (as3_int*)&Ashf[am][ak + t],
                    0, 0);
        }
        __builtin_amdgcn_s_wait_asynccnt(0);
        if (!chunkFull) {                        // zero ragged slots (rare)
#pragma unroll
            for (int t = 0; t < 4; ++t) {
                int k = kk + ak + t;
                if (!(rowv && k < KK)) Ashf[am][ak + t] = 0.f;
            }
        }
#else
        // ---- synchronous fallback: batch loads, then store to LDS ----
        {
            float av[4]; bool aok[4];
#pragma unroll
            for (int t = 0; t < 4; ++t) {
                int k = kk + ak + t;
                aok[t] = rowv && (k < KK);
                av[t]  = w[aok[t] ? wrow + (unsigned)k : 0u];
            }
#pragma unroll
            for (int t = 0; t < 4; ++t)
                Ashf[am][ak + t] = aok[t] ? av[t] : 0.f;
        }
#endif
        // prefetch next weight chunk while this one is consumed
        {
            int nk = kk + 32 + ak;
            if (rowv && nk < KK) __builtin_prefetch(&w[wrow + (unsigned)nk], 0, 3);
        }
        __syncthreads();

        // ---- A fragment from LDS: four 16-byte ds reads + f32->f16 ----
        v16h a;
        {
            const v4f r0 = *(const v4f*)&Ashf[lo][hi * 8];
            const v4f r1 = *(const v4f*)&Ashf[lo][hi * 8 + 4];
            const v4f r2 = *(const v4f*)&Ashf[lo][16 + hi * 8];
            const v4f r3 = *(const v4f*)&Ashf[lo][16 + hi * 8 + 4];
#pragma unroll
            for (int j = 0; j < 4; ++j) {
                a[j]      = (_Float16)r0[j];
                a[4 + j]  = (_Float16)r1[j];
                a[8 + j]  = (_Float16)r2[j];
                a[12 + j] = (_Float16)r3[j];
            }
        }

        // ---- B fragment: two-phase branchless gather (16 loads in flight) ----
        v16h b;
        {
            float tv[16]; bool okb[16];
            const int kb0 = kk + hi * 8;
#pragma unroll
            for (int j = 0; j < 16; ++j) {
                int kb  = kb0 + j + (j >= 8 ? 8 : 0);
                int ci  = kb / (KSZ * KSZ);
                int rr  = kb - ci * (KSZ * KSZ);
                int kh  = rr / KSZ;
                int kw2 = rr - kh * KSZ;
                int ih  = oh - pad + kh;
                int iw  = ow - pad + kw2;
                bool ok = colv && (kb < KK) &&
                          ((unsigned)ih < (unsigned)H) && ((unsigned)iw < (unsigned)W);
                okb[j]  = ok;
                unsigned idx = ok ? (unsigned)(((xbase + ci) * H + ih) * W + iw) : 0u;
                tv[j] = x[idx];                  // independent, pipelined loads
            }
#pragma unroll
            for (int j = 0; j < 16; ++j)
                b[j] = (_Float16)(okb[j] ? tv[j] : 0.f);
        }

        acc = __builtin_amdgcn_wmma_f32_16x16x32_f16(false, a, false, b,
                                                     (short)0, acc, false, false);
        __syncthreads();                         // Ashf reused next iteration
    }

    // ---- epilogue: hoisted bias, wave-uniform full-tile fast path ----
    float bsv[8];
#pragma unroll
    for (int r = 0; r < 8; ++r) {
        int m = mBase + r + hi * 8;
        bsv[r] = bias[m < Cout ? m : 0];
    }
    const unsigned ybase = (unsigned)(bN * Cout) * (unsigned)OHW + (unsigned)rP;
    const bool fullM = (mBase + 16 <= Cout);
    const bool fullN = (nTile * 16 + 16 <= P);   // wave-uniform
    if (fullM && fullN) {
#pragma unroll
        for (int r = 0; r < 8; ++r) {
            unsigned m = (unsigned)(mBase + r + hi * 8);
            y[ybase + m * (unsigned)OHW] = acc[r] + bsv[r];
        }
    } else {
#pragma unroll
        for (int r = 0; r < 8; ++r) {
            int m = mBase + r + hi * 8;
            if (m < Cout && colv)
                y[ybase + (unsigned)m * (unsigned)OHW] = acc[r] + bsv[r];
        }
    }
}

// ---------------------------------------------------------------------------
// FC: D[M x N] = A[M x K] * B[K x N] + bias[N]   (latent @ fc_w + fc_b)
// ---------------------------------------------------------------------------
__global__ void fc_wmma_kernel(const float* __restrict__ A, const float* __restrict__ Bm,
                               const float* __restrict__ bias, float* __restrict__ D,
                               int M, int N, int K) {
    const int lane  = threadIdx.x & 31;
    const int wave  = threadIdx.x >> 5;
    const int nTile = blockIdx.x * 4 + wave;
    const int hi    = lane >> 4;
    const int lo    = lane & 15;
    const int mBase = blockIdx.y * 16;
    const int col   = nTile * 16 + lo;
    const bool colv = (col < N);
    const unsigned ccol = (unsigned)(colv ? col : 0);
    const int mrow  = mBase + lo;
    const bool mv   = (mrow < M);
    const float biasv = bias[ccol];              // hoisted (same for all r)

    v8f acc = {};
    for (int kk = 0; kk < K; kk += 32) {
        v16h a, b;
        {
            float ta[16], tb[16]; bool oka[16], okb[16];
#pragma unroll
            for (int j = 0; j < 16; ++j) {
                int kaBase = (j < 8) ? j : (j + 8);
                int ka = kk + kaBase + (hi ? 8 : 0);
                oka[j] = mv && (ka < K);
                ta[j]  = A[oka[j] ? (unsigned)(mrow * K + ka) : 0u];
                int kb = kk + hi * 8 + j + (j >= 8 ? 8 : 0);
                okb[j] = colv && (kb < K);
                tb[j]  = Bm[okb[j] ? (unsigned)kb * (unsigned)N + ccol : 0u];
            }
#pragma unroll
            for (int j = 0; j < 16; ++j) {
                a[j] = (_Float16)(oka[j] ? ta[j] : 0.f);
                b[j] = (_Float16)(okb[j] ? tb[j] : 0.f);
            }
        }
        acc = __builtin_amdgcn_wmma_f32_16x16x32_f16(false, a, false, b,
                                                     (short)0, acc, false, false);
    }
    const bool fullM = (mBase + 16 <= M);
    const bool fullN = (nTile * 16 + 16 <= N);
    if (fullM && fullN) {
#pragma unroll
        for (int r = 0; r < 8; ++r) {
            unsigned m = (unsigned)(mBase + r + hi * 8);
            D[m * (unsigned)N + (unsigned)col] = acc[r] + biasv;
        }
    } else {
#pragma unroll
        for (int r = 0; r < 8; ++r) {
            int m = mBase + r + hi * 8;
            if (m < M && colv)
                D[(unsigned)m * (unsigned)N + (unsigned)col] = acc[r] + biasv;
        }
    }
}

// ---------------------------------------------------------------------------
// BatchNorm (training mode): stats over (N,H,W) per channel, then fused
// normalize + affine + LeakyReLU in place.
// ---------------------------------------------------------------------------
__global__ void bn_stats_kernel(const float* __restrict__ x, float* __restrict__ mean,
                                float* __restrict__ var, int C, int HW) {
    const int c = blockIdx.x;
    float s = 0.f, ss = 0.f;
    for (int b = 0; b < BATCH; ++b) {
        const float* p = x + (size_t)(b * C + c) * HW;
        for (int i = threadIdx.x; i < HW; i += blockDim.x) {
            float v = p[i];
            s += v; ss += v * v;
        }
    }
    __shared__ float shs[256], shq[256];
    shs[threadIdx.x] = s; shq[threadIdx.x] = ss;
    __syncthreads();
    for (int t = 128; t > 0; t >>= 1) {
        if ((int)threadIdx.x < t) {
            shs[threadIdx.x] += shs[threadIdx.x + t];
            shq[threadIdx.x] += shq[threadIdx.x + t];
        }
        __syncthreads();
    }
    if (threadIdx.x == 0) {
        float cnt = (float)(BATCH * HW);
        float m = shs[0] / cnt;
        mean[c] = m;
        var[c]  = shq[0] / cnt - m * m;          // biased, like jnp.var
    }
}

// 2D grid: blockIdx.y = plane (b*C + c); channel decode in SALU, not per lane
__global__ void bn_lrelu_kernel(float* __restrict__ x, const float* __restrict__ mean,
                                const float* __restrict__ var, const float* __restrict__ g,
                                const float* __restrict__ b, int C, int HW) {
    const int plane = blockIdx.y;
    const int c = plane % C;
    const int i = blockIdx.x * blockDim.x + threadIdx.x;
    if (i >= HW) return;
    const float sc = rsqrtf(var[c] + BN_EPS) * g[c];
    const float sh = b[c] - mean[c] * sc;
    size_t off = (size_t)plane * HW + i;
    float v = x[off] * sc + sh;
    x[off] = (v >= 0.f) ? v : 0.2f * v;
}

// ---------------------------------------------------------------------------
// MaxPool2d(kernel=2, stride=2, padding=1): OH = H/2 + 1
// ---------------------------------------------------------------------------
__global__ void maxpool_kernel(const float* __restrict__ x, float* __restrict__ y,
                               int BC, int H, int W, int OH, int OW) {
    int i = blockIdx.x * blockDim.x + threadIdx.x;
    int total = BC * OH * OW;
    if (i >= total) return;
    int ow = i % OW; int t = i / OW;
    int oh = t % OH; int bc = t / OH;
    float m = -3.4e38f;
    int ih0 = oh * 2 - 1, iw0 = ow * 2 - 1;
#pragma unroll
    for (int dy = 0; dy < 2; ++dy)
#pragma unroll
        for (int dx = 0; dx < 2; ++dx) {
            int ih = ih0 + dy, iw = iw0 + dx;
            bool ok = ((unsigned)ih < (unsigned)H) && ((unsigned)iw < (unsigned)W);
            float v = x[ok ? (unsigned)((bc * H + ih) * W + iw) : 0u];
            v = ok ? v : -3.4e38f;
            m = (m > v) ? m : v;
        }
    y[i] = m;
}

// Nearest upsample by f, writing into channel slice [cOff, cOff+C) of a
// Cdst-channel destination (fuses the concat for free).
__global__ void upsample_cat_kernel(const float* __restrict__ x, float* __restrict__ y,
                                    int C, int H, int W, int f, int Cdst, int cOff,
                                    int total) {
    int i = blockIdx.x * blockDim.x + threadIdx.x;
    if (i >= total) return;                     // total = BATCH*C*(H*f)*(W*f)
    const int OW = W * f, OH = H * f;
    int ow = i % OW; int t = i / OW;
    int oh = t % OH; t /= OH;
    int c  = t % C;  int b = t / C;
    y[((b * Cdst + cOff + c) * OH + oh) * OW + ow] =
        x[((b * C + c) * H + oh / f) * W + ow / f];
}

// Per-channel mean / unbiased std of x4[0] (256 channels x 64 elems each)
__global__ void slab_stats_kernel(const float* __restrict__ x4,
                                  float* __restrict__ sm, float* __restrict__ sd) {
    int c = blockIdx.x * blockDim.x + threadIdx.x;
    if (c >= 256) return;
    const float* p = x4 + c * 64;               // batch element 0
    float s = 0.f;
    for (int i = 0; i < 64; ++i) s += p[i];
    float m = s * (1.0f / 64.0f);
    float ss = 0.f;
    for (int i = 0; i < 64; ++i) { float d = p[i] - m; ss += d * d; }
    sm[c] = m;
    sd[c] = sqrtf(ss * (1.0f / 63.0f));         // ddof=1 (torch-style)
}

// var_c = mean + std * eps, eps deterministic pseudo-normal, same across batch
__global__ void var_fill_kernel(const float* __restrict__ sm, const float* __restrict__ sd,
                                float* __restrict__ varC, int total) {
    int i = blockIdx.x * blockDim.x + threadIdx.x;
    if (i >= total) return;
    int r = i & 16383;                          // position within [256,8,8]
    int c = r >> 6;
    unsigned h1 = hash_u32(2u * (unsigned)r + 1u);
    unsigned h2 = hash_u32(2u * (unsigned)r + 2u);
    float u1 = ((float)h1 + 0.5f) * 2.3283064e-10f;
    float u2 = ((float)h2 + 0.5f) * 2.3283064e-10f;
    float z  = sqrtf(-2.0f * logf(u1)) * cosf(6.2831853f * u2);
    varC[i] = sm[c] + sd[c] * z;
}

__global__ void tanh_kernel(const float* __restrict__ x, float* __restrict__ y, int total) {
    int i = blockIdx.x * blockDim.x + threadIdx.x;
    if (i >= total) return;
    y[i] = tanhf(x[i]);
}

// ---------------------------------------------------------------------------
// Host orchestration
// ---------------------------------------------------------------------------
extern "C" void kernel_launch(void* const* d_in, const int* in_sizes, int n_in,
                              void* d_out, int out_size, void* d_ws, size_t ws_size,
                              hipStream_t stream) {
    (void)in_sizes; (void)n_in; (void)out_size; (void)ws_size;

    const float* in_mask = (const float*)d_in[0];   // [64,1,128,128]
    const float* latent  = (const float*)d_in[1];   // [64,100]
    // params flattened in dict order: c{i}_w, c{i}_b (i=1..18) -> idx 2..37,
    // bn{i}_g, bn{i}_b (i=1..18) -> idx 38..73, fc_w=74, fc_b=75.
    auto Wc = [&](int i) { return (const float*)d_in[2 + 2 * i]; };
    auto Bc = [&](int i) { return (const float*)d_in[3 + 2 * i]; };
    auto Gn = [&](int i) { return (const float*)d_in[38 + 2 * i]; };
    auto Bn = [&](int i) { return (const float*)d_in[39 + 2 * i]; };
    const float* fcw = (const float*)d_in[74];
    const float* fcb = (const float*)d_in[75];

    size_t off = 0;
    auto alloc = [&](size_t nfloats) {
        float* p = (float*)((char*)d_ws + off);
        off += (nfloats * sizeof(float) + 255) & ~(size_t)255;
        return p;
    };

    float* bnM  = alloc(256);
    float* bnV  = alloc(256);
    float* sm   = alloc(256);
    float* sd   = alloc(256);
    float* a1   = alloc((size_t)64 * 16 * 126 * 126);
    float* hbuf = alloc((size_t)64 * 16 * 64 * 64);
    float* a2   = alloc((size_t)64 * 32 * 62 * 62);
    float* x1   = alloc((size_t)64 * 32 * 32 * 32);
    float* a3   = alloc((size_t)64 * 64 * 30 * 30);
    float* x2   = alloc((size_t)64 * 64 * 16 * 16);
    float* a4   = alloc((size_t)64 * 128 * 14 * 14);
    float* x3   = alloc((size_t)64 * 128 * 8 * 8);
    float* x4   = alloc((size_t)64 * 256 * 8 * 8);
    float* varC = alloc((size_t)64 * 256 * 8 * 8);
    float* n1   = alloc((size_t)64 * 256 * 8 * 8);
    float* t7   = alloc((size_t)64 * 128 * 8 * 8);
    float* t8   = alloc((size_t)64 * 64 * 8 * 8);
    float* t6   = alloc((size_t)64 * 64 * 8 * 8);
    float* y1   = alloc((size_t)64 * 128 * 16 * 16);
    float* t9   = alloc((size_t)64 * 64 * 16 * 16);
    float* t10  = alloc((size_t)64 * 32 * 16 * 16);
    float* t11  = alloc((size_t)64 * 32 * 8 * 8);
    float* y2   = alloc((size_t)64 * 64 * 32 * 32);
    float* t12  = alloc((size_t)64 * 32 * 32 * 32);
    float* t13  = alloc((size_t)64 * 16 * 32 * 32);
    float* t14  = alloc((size_t)64 * 16 * 16 * 16);
    float* y3   = alloc((size_t)64 * 32 * 64 * 64);
    float* t15  = alloc((size_t)64 * 16 * 64 * 64);
    float* t16  = alloc((size_t)64 * 8 * 64 * 64);
    float* t17  = alloc((size_t)64 * 8 * 32 * 32);
    float* y4   = alloc((size_t)64 * 16 * 128 * 128);
    float* pre  = alloc((size_t)64 * 1 * 128 * 128);

    auto conv = [&](int ci, const float* in, float* out,
                    int Cin, int H, int W, int Cout, int pad, int K) {
        int OH = H + 2 * pad - K + 1, OW = W + 2 * pad - K + 1;
        int P = BATCH * OH * OW;
        int ntiles = (P + 15) / 16;
        dim3 grid((ntiles + 3) / 4, (Cout + 15) / 16);
        if (K == 3)
            conv2d_wmma_kernel<3><<<grid, 128, 0, stream>>>(
                in, Wc(ci), Bc(ci), out, Cin, H, W, Cout, OH, OW, pad);
        else
            conv2d_wmma_kernel<1><<<grid, 128, 0, stream>>>(
                in, Wc(ci), Bc(ci), out, Cin, H, W, Cout, OH, OW, pad);
    };
    auto bn = [&](int bi, float* buf, int C, int HW) {
        bn_stats_kernel<<<C, 256, 0, stream>>>(buf, bnM, bnV, C, HW);
        dim3 grid((HW + 255) / 256, BATCH * C);
        bn_lrelu_kernel<<<grid, 256, 0, stream>>>(buf, bnM, bnV, Gn(bi), Bn(bi), C, HW);
    };
    auto pool = [&](const float* in, float* out, int BC, int H, int W) {
        int OH = H / 2 + 1, OW = W / 2 + 1;
        int total = BC * OH * OW;
        maxpool_kernel<<<(total + 255) / 256, 256, 0, stream>>>(in, out, BC, H, W, OH, OW);
    };
    auto up = [&](const float* in, float* out, int C, int H, int W, int f,
                  int Cdst, int cOff) {
        int total = BATCH * C * H * f * W * f;
        upsample_cat_kernel<<<(total + 255) / 256, 256, 0, stream>>>(
            in, out, C, H, W, f, Cdst, cOff, total);
    };

    // ---------------- encoder ----------------
    conv(0, in_mask, a1, 1, 128, 128, 16, 0, 3);  bn(0, a1, 16, 126 * 126);
    pool(a1, hbuf, 64 * 16, 126, 126);                       // [64,16,64,64]
    conv(1, hbuf, a2, 16, 64, 64, 32, 0, 3);      bn(1, a2, 32, 62 * 62);
    pool(a2, x1, 64 * 32, 62, 62);                           // [64,32,32,32]
    conv(2, x1, a3, 32, 32, 32, 64, 0, 3);        bn(2, a3, 64, 30 * 30);
    pool(a3, x2, 64 * 64, 30, 30);                           // [64,64,16,16]
    conv(3, x2, a4, 64, 16, 16, 128, 0, 3);       bn(3, a4, 128, 14 * 14);
    pool(a4, x3, 64 * 128, 14, 14);                          // [64,128,8,8]
    conv(4, x3, x4, 128, 8, 8, 256, 1, 3);        bn(4, x4, 256, 64);  // [64,256,8,8]

    // ---------------- variational resample ----------------
    slab_stats_kernel<<<1, 256, 0, stream>>>(x4, sm, sd);
    var_fill_kernel<<<(64 * 256 * 64 + 255) / 256, 256, 0, stream>>>(
        sm, sd, varC, 64 * 256 * 64);

    // ---------------- noise branch ----------------
    {
        int M = 64, N = 256 * 64, K = 100;
        int ntiles = (N + 15) / 16;
        dim3 grid((ntiles + 3) / 4, (M + 15) / 16);
        fc_wmma_kernel<<<grid, 128, 0, stream>>>(latent, fcw, fcb, n1, M, N, K);
    }
    bn(6, n1, 256, 64);                                       // bn7
    conv(6, n1, t7, 256, 8, 8, 128, 1, 3);        bn(7, t7, 128, 64);   // c7/bn8
    conv(7, t7, t8, 128, 8, 8, 64, 1, 3);         bn(8, t8, 64, 64);    // c8/bn9
    up(t8, y1, 64, 8, 8, 2, 128, 64);                         // n2 -> y1[:,64:128]
    conv(5, varC, t6, 256, 8, 8, 64, 1, 3);       bn(5, t6, 64, 64);    // c6/bn6
    up(t6, y1, 64, 8, 8, 2, 128, 0);                          // v_out -> y1[:,0:64]

    // ---------------- decoder ----------------
    conv(8, y1, t9, 128, 16, 16, 64, 1, 3);       bn(9, t9, 64, 256);    // c9/bn10
    conv(9, t9, t10, 64, 16, 16, 32, 1, 3);       bn(10, t10, 32, 256);  // c10/bn11
    up(t10, y2, 32, 16, 16, 2, 64, 32);                       // tmp1 -> y2[:,32:64]
    conv(10, x3, t11, 128, 8, 8, 32, 0, 1);       bn(11, t11, 32, 64);   // c11/bn12
    up(t11, y2, 32, 8, 8, 4, 64, 0);                          // u1 -> y2[:,0:32]

    conv(11, y2, t12, 64, 32, 32, 32, 1, 3);      bn(12, t12, 32, 1024); // c12/bn13
    conv(12, t12, t13, 32, 32, 32, 16, 1, 3);     bn(13, t13, 16, 1024); // c13/bn14
    up(t13, y3, 16, 32, 32, 2, 32, 16);                       // tmp2 -> y3[:,16:32]
    conv(13, x2, t14, 64, 16, 16, 16, 0, 1);      bn(14, t14, 16, 256);  // c14/bn15
    up(t14, y3, 16, 16, 16, 4, 32, 0);                        // u2 -> y3[:,0:16]

    conv(14, y3, t15, 32, 64, 64, 16, 1, 3);      bn(15, t15, 16, 4096); // c15/bn16
    conv(15, t15, t16, 16, 64, 64, 8, 1, 3);      bn(16, t16, 8, 4096);  // c16/bn17
    up(t16, y4, 8, 64, 64, 2, 16, 8);                         // tmp3 -> y4[:,8:16]
    conv(16, x1, t17, 32, 32, 32, 8, 0, 1);       bn(17, t17, 8, 1024);  // c17/bn18
    up(t17, y4, 8, 32, 32, 4, 16, 0);                         // u3 -> y4[:,0:8]

    // ---------------- head ----------------
    conv(17, y4, pre, 16, 128, 128, 1, 0, 1);                 // c18 (no BN)
    tanh_kernel<<<(64 * 128 * 128 + 255) / 256, 256, 0, stream>>>(
        pre, (float*)d_out, 64 * 128 * 128);
}
